// NodeModelBP_50242527429369
// MI455X (gfx1250) — matrix operand order repaired and checked by reading, not compiled, not run on hardware
//
#include <hip/hip_runtime.h>

#define N_NODES 50000
#define N_EDGES 800000
#define D_X     128
#define D_LSTM  128
#define D_Z     64
#define D_E     96
#define D_IN    416   // 13 * 32
#define D_HID   512   // 16 * 32, 32 * 16
#define D_OUT   256   // 16 * 16

typedef __attribute__((ext_vector_type(16))) __bf16 v16bf;
typedef __attribute__((ext_vector_type(8)))  __bf16 v8bf;
typedef __attribute__((ext_vector_type(8)))  float  v8f;

// ---------------------------------------------------------------------------
// CDNA5 async global->LDS copy helpers (ASYNCcnt-tracked, bypasses VGPRs)
// GVS addressing: mem = SGPR64 + VADDR(i32); LDS dest = VDST VGPR byte offset.
// ---------------------------------------------------------------------------
__device__ __forceinline__ void async_ld_b128(unsigned lds_addr,
                                              const void* sbase,
                                              unsigned voff) {
    asm volatile("global_load_async_to_lds_b128 %0, %1, %2"
                 :: "v"(lds_addr), "v"(voff), "s"(sbase) : "memory");
}
__device__ __forceinline__ void wait_async0() {
    asm volatile("s_wait_asynccnt 0" ::: "memory");
}

// ---------------------------------------------------------------------------
// Kernel 1: zero the agg scratch buffer (harness poisons ws with 0xAA)
// ---------------------------------------------------------------------------
__global__ void zero_agg_kernel(float* __restrict__ agg) {
    int i = blockIdx.x * blockDim.x + threadIdx.x;
    if (i < N_NODES * D_E) agg[i] = 0.0f;
}

// ---------------------------------------------------------------------------
// Kernel 2: repack W1 / W2 (f32, row-major [K,N]) into bf16 WMMA B-fragment
// order.  Layout: t = ((kt*NT + nt)*32 + lane)*16 + j
//   element j of lane's v16bf  =  W[kt*32 + (lane<16?0:8) + j + (j<8?0:8)][nt*16 + lane%16]
// so a B tile is one contiguous 32B load per lane.
// ---------------------------------------------------------------------------
__global__ void repack_weights_kernel(const float* __restrict__ W1,
                                      const float* __restrict__ W2,
                                      __bf16* __restrict__ W1p,
                                      __bf16* __restrict__ W2p) {
    int t = blockIdx.x * blockDim.x + threadIdx.x;
    const int n1 = D_IN * D_HID;   // 212992, 13 kt * 32 nt
    const int n2 = D_HID * D_OUT;  // 131072, 16 kt * 16 nt
    if (t < n1) {
        int j  = t & 15;
        int l  = (t >> 4) & 31;
        int nt = (t >> 9) & 31;        // 32 nt tiles per kt (512 elems per nt)
        int kt = t >> 14;              // 16384 elems per kt
        int k  = kt * 32 + ((l < 16) ? 0 : 8) + j + ((j < 8) ? 0 : 8);
        int n  = nt * 16 + (l & 15);
        W1p[t] = (__bf16)W1[k * D_HID + n];
    } else if (t < n1 + n2) {
        int u  = t - n1;
        int j  = u & 15;
        int l  = (u >> 4) & 31;
        int nt = (u >> 9) & 15;        // 16 nt tiles per kt
        int kt = u >> 13;              // 8192 elems per kt
        int k  = kt * 32 + ((l < 16) ? 0 : 8) + j + ((j < 8) ? 0 : 8);
        int n  = nt * 16 + (l & 15);
        W2p[u] = (__bf16)W2[k * D_OUT + n];
    }
}

// ---------------------------------------------------------------------------
// Kernel 3: scatter-sum edge features into agg (L2-resident destination)
// ---------------------------------------------------------------------------
__global__ void scatter_edges_kernel(const float* __restrict__ edge_attr,
                                     const int* __restrict__ edge_row,
                                     float* __restrict__ agg) {
    int t = blockIdx.x * blockDim.x + threadIdx.x;   // 76.8M < 2^31
    if (t < N_EDGES * D_E) {
        int e = t / D_E;
        int j = t - e * D_E;
        int r = edge_row[e];                          // row = edge_index[0]
        atomicAdd(&agg[r * D_E + j], edge_attr[t]);
    }
}

// ---------------------------------------------------------------------------
// Kernel 4: fused  relu(concat @ W1 + b1) @ W2 + b2
// 256 threads = 8 waves; each block does a 128-row M tile, each wave a 16-row
// strip.  Weight k-panels are double-buffered in LDS and prefetched with
// global_load_async_to_lds_b128 while WMMAs consume the current panel.
// h strip stays in LDS (bf16) between the two GEMMs.
// ---------------------------------------------------------------------------
__launch_bounds__(256)
__global__ void fused_mlp_kernel(const float* __restrict__ x,
                                 const float* __restrict__ xl,
                                 const float* __restrict__ z,
                                 const float* __restrict__ agg,
                                 const __bf16* __restrict__ W1p,
                                 const float* __restrict__ b1,
                                 const __bf16* __restrict__ W2p,
                                 const float* __restrict__ b2,
                                 float* __restrict__ out) {
    __shared__ alignas(32) __bf16 panel[2][32 * 512];  // 2 x 32 KB panels
    __shared__ alignas(32) __bf16 hbuf[128 * 512];     // 128 KB (h, bf16)

    const int tid  = threadIdx.x;
    const int wave = tid >> 5;
    const int lane = tid & 31;
    const int mrow = lane & 15;
    const int half = lane >> 4;                // 0: lanes 0-15, 1: lanes 16-31
    const int m0   = blockIdx.x * 128 + wave * 16;
    int arow = m0 + mrow;
    if (arow >= N_NODES) arow = N_NODES - 1;   // clamp loads; stores guarded

    const unsigned panel_lds[2] = {
        (unsigned)(size_t)(void*)&panel[0][0],
        (unsigned)(size_t)(void*)&panel[1][0]
    };

    // ---------------- GEMM1: h = relu(concat @ W1 + b1) -------------------
    v8f acc[32];
    #pragma unroll
    for (int nt = 0; nt < 32; ++nt) {
        float bv = b1[nt * 16 + mrow];
        #pragma unroll
        for (int v = 0; v < 8; ++v) acc[nt][v] = bv;
    }

    // prefetch panel for kt = 0 into buffer 0: 32 KB = 2048 x 16B, 8 per thread
    #pragma unroll
    for (int i = 0; i < 8; ++i) {
        unsigned off = (unsigned)(tid + i * 256) * 16u;
        async_ld_b128(panel_lds[0] + off, (const void*)W1p, off);
    }

    for (int kt = 0; kt < 13; ++kt) {
        wait_async0();        // own async slice of panel[kt&1] landed in LDS
        __syncthreads();      // whole panel complete; prev buffer free

        if (kt + 1 < 13) {    // prefetch next panel into the other buffer
            const void* src = (const void*)(W1p + (kt + 1) * 16384);
            #pragma unroll
            for (int i = 0; i < 8; ++i) {
                unsigned off = (unsigned)(tid + i * 256) * 16u;
                async_ld_b128(panel_lds[(kt + 1) & 1] + off, src, off);
            }
        }

        // A fragment: virtual concat — pick source segment for this k-tile
        const float* srcp; int ld; int col0;
        if (kt < 4)       { srcp = x;   ld = D_X;    col0 = kt * 32; }
        else if (kt < 8)  { srcp = xl;  ld = D_LSTM; col0 = (kt - 4) * 32; }
        else if (kt < 10) { srcp = z;   ld = D_Z;    col0 = (kt - 8) * 32; }
        else              { srcp = agg; ld = D_E;    col0 = (kt - 10) * 32; }

        const float* rp = srcp + (size_t)arow * ld + col0 + half * 8;
        float4 c0 = *(const float4*)(rp);
        float4 c1 = *(const float4*)(rp + 4);
        float4 c2 = *(const float4*)(rp + 16);
        float4 c3 = *(const float4*)(rp + 20);
        v16bf a;
        a[0] = (__bf16)c0.x;  a[1] = (__bf16)c0.y;  a[2] = (__bf16)c0.z;  a[3] = (__bf16)c0.w;
        a[4] = (__bf16)c1.x;  a[5] = (__bf16)c1.y;  a[6] = (__bf16)c1.z;  a[7] = (__bf16)c1.w;
        a[8] = (__bf16)c2.x;  a[9] = (__bf16)c2.y;  a[10] = (__bf16)c2.z; a[11] = (__bf16)c2.w;
        a[12] = (__bf16)c3.x; a[13] = (__bf16)c3.y; a[14] = (__bf16)c3.z; a[15] = (__bf16)c3.w;

        const __bf16* pan = &panel[kt & 1][0];
        #pragma unroll
        for (int nt = 0; nt < 32; ++nt) {
            v16bf b = *(const v16bf*)(pan + nt * 512 + lane * 16);
            acc[nt] = __builtin_amdgcn_wmma_f32_16x16x32_bf16(
                false, a, false, b, (short)0, acc[nt], false, false);
        }
    }

    // relu -> bf16 h strip in LDS (wave-private region, row-major 16x512)
    #pragma unroll
    for (int nt = 0; nt < 32; ++nt) {
        #pragma unroll
        for (int v = 0; v < 8; ++v) {
            float hv = acc[nt][v];
            hv = hv > 0.0f ? hv : 0.0f;
            hbuf[(wave * 16 + v + 8 * half) * 512 + nt * 16 + mrow] = (__bf16)hv;
        }
    }

    // all waves done with panel[0] (kt=12 used it) before GEMM2 prefetch
    __syncthreads();

    // ---------------- GEMM2: out = h @ W2 + b2 ----------------------------
    v8f acc2[16];
    #pragma unroll
    for (int nt = 0; nt < 16; ++nt) {
        float bv = b2[nt * 16 + mrow];
        #pragma unroll
        for (int v = 0; v < 8; ++v) acc2[nt][v] = bv;
    }

    // prefetch W2 panel kt=0 into buffer 0: 16 KB = 1024 x 16B, 4 per thread
    #pragma unroll
    for (int i = 0; i < 4; ++i) {
        unsigned off = (unsigned)(tid + i * 256) * 16u;
        async_ld_b128(panel_lds[0] + off, (const void*)W2p, off);
    }

    for (int kt = 0; kt < 16; ++kt) {
        wait_async0();
        __syncthreads();

        if (kt + 1 < 16) {
            const void* src = (const void*)(W2p + (kt + 1) * 8192);
            #pragma unroll
            for (int i = 0; i < 4; ++i) {
                unsigned off = (unsigned)(tid + i * 256) * 16u;
                async_ld_b128(panel_lds[(kt + 1) & 1] + off, src, off);
            }
        }

        // A fragment of h from LDS: two contiguous 16B reads per lane
        const __bf16* hrow = hbuf + (wave * 16 + mrow) * 512 + kt * 32 + half * 8;
        v8bf lo = *(const v8bf*)(hrow);
        v8bf hi = *(const v8bf*)(hrow + 16);
        v16bf a2 = __builtin_shufflevector(lo, hi,
            0, 1, 2, 3, 4, 5, 6, 7, 8, 9, 10, 11, 12, 13, 14, 15);

        const __bf16* pan = &panel[kt & 1][0];
        #pragma unroll
        for (int nt = 0; nt < 16; ++nt) {
            v16bf b = *(const v16bf*)(pan + nt * 512 + lane * 16);
            acc2[nt] = __builtin_amdgcn_wmma_f32_16x16x32_bf16(
                false, a2, false, b, (short)0, acc2[nt], false, false);
        }
    }

    // store (f32), guard M boundary
    #pragma unroll
    for (int nt = 0; nt < 16; ++nt) {
        #pragma unroll
        for (int v = 0; v < 8; ++v) {
            int row = m0 + v + 8 * half;
            if (row < N_NODES)
                out[(size_t)row * D_OUT + nt * 16 + mrow] = acc2[nt][v];
        }
    }
}

// ---------------------------------------------------------------------------
extern "C" void kernel_launch(void* const* d_in, const int* in_sizes, int n_in,
                              void* d_out, int out_size, void* d_ws, size_t ws_size,
                              hipStream_t stream) {
    const float* x   = (const float*)d_in[0];
    const float* xl  = (const float*)d_in[1];
    const float* z   = (const float*)d_in[2];
    const float* ea  = (const float*)d_in[3];
    const int*   ei  = (const int*)d_in[4];   // [2, N_EDGES]; row = first half
    const float* W1  = (const float*)d_in[5];
    const float* b1  = (const float*)d_in[6];
    const float* W2  = (const float*)d_in[7];
    const float* b2  = (const float*)d_in[8];
    float* out = (float*)d_out;

    char* ws = (char*)d_ws;
    float*  agg = (float*)ws;                                   // 19,200,000 B
    __bf16* W1p = (__bf16*)(ws + 19200000);                     //    425,984 B
    __bf16* W2p = (__bf16*)(ws + 19200000 + 425984);            //    262,144 B

    zero_agg_kernel<<<(N_NODES * D_E + 255) / 256, 256, 0, stream>>>(agg);
    repack_weights_kernel<<<(D_IN * D_HID + D_HID * D_OUT + 255) / 256, 256, 0, stream>>>(
        W1, W2, W1p, W2p);
    scatter_edges_kernel<<<(N_EDGES * D_E + 255) / 256, 256, 0, stream>>>(ea, ei, agg);
    fused_mlp_kernel<<<(N_NODES + 127) / 128, 256, 0, stream>>>(
        x, xl, z, agg, W1p, b1, W2p, b2, out);
}
